// BeansBackboneV2_40948218200754
// MI455X (gfx1250) — matrix operand order repaired and checked by reading, not compiled, not run on hardware
//
#include <hip/hip_runtime.h>
#include <cstdint>
#include <cstddef>

typedef _Float16 f16;
typedef __attribute__((ext_vector_type(16))) _Float16 v16h;
typedef __attribute__((ext_vector_type(8)))  _Float16 v8h;
typedef __attribute__((ext_vector_type(8)))  float    v8f;
typedef __attribute__((ext_vector_type(2)))  float    v2f;
typedef __attribute__((ext_vector_type(4)))  int      v4i;

constexpr int Bn = 2, Sn = 1025, Dn = 1024, Hn = 16, Pn = 1024, Kn = 32, HDn = 64;
constexpr float TEMPER = 0.1f;
constexpr float SCALEQ = 0.125f;      // HD^-0.5 = 64^-0.5
constexpr float EPSLN  = 1e-5f;

#ifndef __has_builtin
#define __has_builtin(x) 0
#endif

#if defined(__gfx1250__) && __has_builtin(__builtin_amdgcn_global_load_async_to_lds_b128) && __has_builtin(__builtin_amdgcn_s_wait_asynccnt)
#define HAVE_ASYNC_LDS 1
#else
#define HAVE_ASYNC_LDS 0
#endif

#if HAVE_ASYNC_LDS
// builtin signature: (int4 AS1* src, int4 AS3* dst, imm offset, imm cpol)
typedef __attribute__((address_space(1))) v4i gv4i;
typedef __attribute__((address_space(3))) v4i lv4i;
#endif

// ---------------------------------------------------------------------------
// f32 -> f16 convert
// ---------------------------------------------------------------------------
__global__ void cvt_f16_k(const float* __restrict__ in, f16* __restrict__ out, long n) {
    long i = (long)blockIdx.x * blockDim.x + threadIdx.x;
    long stride = (long)gridDim.x * blockDim.x;
    for (; i < n; i += stride) out[i] = (f16)in[i];
}

// ---------------------------------------------------------------------------
// LayerNorm: one 256-thread block per row of D=1024; writes f32 + f16
// ---------------------------------------------------------------------------
__global__ __launch_bounds__(256) void layernorm_k(
    const float* __restrict__ x, const float* __restrict__ w, const float* __restrict__ b,
    float* __restrict__ y, f16* __restrict__ yh)
{
    int row = blockIdx.x, t = threadIdx.x;
    const float* xr = x + (size_t)row * Dn;
    __shared__ float rs[256], rq[256];
    float s = 0.f, sq = 0.f;
    for (int i = t; i < Dn; i += 256) { float v = xr[i]; s += v; sq += v * v; }
    rs[t] = s; rq[t] = sq; __syncthreads();
    for (int o = 128; o > 0; o >>= 1) {
        if (t < o) { rs[t] += rs[t + o]; rq[t] += rq[t + o]; }
        __syncthreads();
    }
    float mean = rs[0] * (1.0f / Dn);
    float var  = rq[0] * (1.0f / Dn) - mean * mean;
    float inv  = rsqrtf(var + EPSLN);
    for (int i = t; i < Dn; i += 256) {
        float v = (xr[i] - mean) * inv * w[i] + b[i];
        size_t o = (size_t)row * Dn + i;
        y[o] = v; yh[o] = (f16)v;
    }
}

// ---------------------------------------------------------------------------
// L2 normalize rows in place (D=1024)
// ---------------------------------------------------------------------------
__global__ __launch_bounds__(256) void l2norm_k(float* __restrict__ y)
{
    int row = blockIdx.x, t = threadIdx.x;
    float* yr = y + (size_t)row * Dn;
    __shared__ float rq[256];
    float sq = 0.f;
    for (int i = t; i < Dn; i += 256) { float v = yr[i]; sq += v * v; }
    rq[t] = sq; __syncthreads();
    for (int o = 128; o > 0; o >>= 1) { if (t < o) rq[t] += rq[t + o]; __syncthreads(); }
    float inv = 1.0f / fmaxf(sqrtf(rq[0]), 1e-12f);
    for (int i = t; i < Dn; i += 256) yr[i] *= inv;
}

// ---------------------------------------------------------------------------
// f16 WMMA GEMM with LDS-staged, double-buffered B tile (async copy if avail):
// C[M,N] = A[M,Kd] @ Wt[N,Kd]^T (+bias)(+act)(+resid)
// Block = 256 threads = 8 waves -> 128 rows x 64 cols. B tile shared via LDS.
// K-loop unrolled by 2 (ping-pong A regs + LDS buffers), last chunk peeled,
// so the steady-state body is branch-free. Requires Kd % 64 == 0, Kd >= 64.
// ---------------------------------------------------------------------------
constexpr int BSTRIDE = 40;   // halves; padded from 32 to spread LDS banks

__device__ __forceinline__ v16h loadA16(const f16* arow, int kk, int hh) {
    v16h a;
    *((v8h*)&a)     = *(const v8h*)(arow + kk + hh * 8);
    *((v8h*)&a + 1) = *(const v8h*)(arow + kk + 16 + hh * 8);
    return a;
}

__device__ __forceinline__ void mma4(const f16* btile0, int l15, int hh,
                                     v16h a, v8f acc[4]) {
    const f16* bp = btile0 + l15 * BSTRIDE + hh * 8;
#pragma unroll
    for (int j = 0; j < 4; ++j) {
        v16h bmat;
        *((v8h*)&bmat)     = *(const v8h*)(bp + j * 16 * BSTRIDE);
        *((v8h*)&bmat + 1) = *(const v8h*)(bp + j * 16 * BSTRIDE + 16);
        acc[j] = __builtin_amdgcn_wmma_f32_16x16x32_f16(false, a, false, bmat,
                                                        (short)0, acc[j], false, false);
    }
}

__global__ __launch_bounds__(256) void wmma_gemm_f16_k(
    const f16* __restrict__ A, const f16* __restrict__ Wt,
    const float* __restrict__ bias, const float* __restrict__ resid,
    float* __restrict__ C, f16* __restrict__ Ch,
    int M, int N, int Kd, int act)
{
    __shared__ f16 btile[2][64 * BSTRIDE];

    const int tid  = threadIdx.x;
    const int wave = tid >> 5;
    const int lane = tid & 31;
    const int l15  = lane & 15;
    const int m0 = blockIdx.x * 128 + wave * 16;
    const int n0 = blockIdx.y * 64;
    const int ml = m0 + l15;
    const int hh = lane >> 4;

    const f16* arow = A + (size_t)(ml < M ? ml : (M - 1)) * Kd;

    // cooperative B copy: thread t -> row br (0..63), 8-half chunk bc
    const int br = tid >> 2;
    const int bc = (tid & 3) * 8;
    const f16* bsrc = Wt + (size_t)(n0 + br) * Kd + bc;
    f16* bdst0 = &btile[0][br * BSTRIDE + bc];
    f16* bdst1 = &btile[1][br * BSTRIDE + bc];

#if HAVE_ASYNC_LDS
    gv4i* gb  = (gv4i*)bsrc;          // AS casts hoisted out of the loop
    lv4i* lb0 = (lv4i*)bdst0;
    lv4i* lb1 = (lv4i*)bdst1;
#endif

    auto syncwait = [&]() {
#if HAVE_ASYNC_LDS
        __builtin_amdgcn_s_wait_asynccnt(0);
#endif
        __syncthreads();
    };

    v8f acc[4] = {};

    // preload: B step 0 -> buffer 0, A step 0 -> regs
#if HAVE_ASYNC_LDS
    __builtin_amdgcn_global_load_async_to_lds_b128(gb, lb0, 0, 0);
#else
    *(v8h*)bdst0 = *(const v8h*)bsrc;
#endif
    v16h a0 = loadA16(arow, 0, hh), a1;

    // steady state: two K-steps per iteration, branch-free bodies
    for (int kk = 0; kk < Kd - 64; kk += 64) {
        syncwait();                                    // buf0 ready
#if HAVE_ASYNC_LDS
        __builtin_amdgcn_global_load_async_to_lds_b128(gb + (kk + 32) / 8, lb1, 0, 0);
#else
        *(v8h*)bdst1 = *(const v8h*)(bsrc + kk + 32);
#endif
        a1 = loadA16(arow, kk + 32, hh);
        mma4(&btile[0][0], l15, hh, a0, acc);

        syncwait();                                    // buf1 ready
#if HAVE_ASYNC_LDS
        __builtin_amdgcn_global_load_async_to_lds_b128(gb + (kk + 64) / 8, lb0, 0, 0);
#else
        *(v8h*)bdst0 = *(const v8h*)(bsrc + kk + 64);
#endif
        a0 = loadA16(arow, kk + 64, hh);
        mma4(&btile[1][0], l15, hh, a1, acc);
    }

    // tail: last 64 of K (prefetch only the final step)
    syncwait();
#if HAVE_ASYNC_LDS
    __builtin_amdgcn_global_load_async_to_lds_b128(gb + (Kd - 32) / 8, lb1, 0, 0);
#else
    *(v8h*)bdst1 = *(const v8h*)(bsrc + Kd - 32);
#endif
    a1 = loadA16(arow, Kd - 32, hh);
    mma4(&btile[0][0], l15, hh, a0, acc);

    syncwait();
    mma4(&btile[1][0], l15, hh, a1, acc);

#pragma unroll
    for (int j = 0; j < 4; ++j) {
#pragma unroll
        for (int r = 0; r < 8; ++r) {
            int row = m0 + r + hh * 8;
            if (row < M) {
                int col = n0 + j * 16 + l15;
                float v = acc[j][r];
                if (bias) v += bias[col];
                if (act == 1) v = 0.5f * v * (1.0f + erff(v * 0.7071067811865475f));
                if (resid) v += resid[(size_t)row * N + col];
                size_t o = (size_t)row * N + col;
                if (C)  C[o]  = v;
                if (Ch) Ch[o] = (f16)v;
            }
        }
    }
}

// ---------------------------------------------------------------------------
// f32 WMMA GEMM (for routing scores, to keep top-k ordering exact):
// C[M,N] = A[M,Kd] @ Bt[N,Kd]^T via v_wmma_f32_16x16x4_f32, wave per 16x64 tile
// ---------------------------------------------------------------------------
__global__ __launch_bounds__(32) void wmma_gemm_f32_k(
    const float* __restrict__ A, const float* __restrict__ Bt,
    float* __restrict__ C, int M, int N, int Kd)
{
    const int lane = threadIdx.x & 31;
    const int m0 = blockIdx.x * 16;
    const int n0 = blockIdx.y * 64;
    const int hh = lane >> 4;

    const float* arow  = A  + (size_t)(m0 + (lane & 15)) * Kd;
    const float* brow0 = Bt + (size_t)(n0 +  0 + (lane & 15)) * Kd;
    const float* brow1 = Bt + (size_t)(n0 + 16 + (lane & 15)) * Kd;
    const float* brow2 = Bt + (size_t)(n0 + 32 + (lane & 15)) * Kd;
    const float* brow3 = Bt + (size_t)(n0 + 48 + (lane & 15)) * Kd;

    v8f acc[4] = {};

    for (int kk = 0; kk < Kd; kk += 4) {
        v2f a  = *(const v2f*)(arow  + kk + hh * 2);
        v2f b0 = *(const v2f*)(brow0 + kk + hh * 2);
        v2f b1 = *(const v2f*)(brow1 + kk + hh * 2);
        v2f b2 = *(const v2f*)(brow2 + kk + hh * 2);
        v2f b3 = *(const v2f*)(brow3 + kk + hh * 2);
        acc[0] = __builtin_amdgcn_wmma_f32_16x16x4_f32(false, a, false, b0, (short)0, acc[0], false, false);
        acc[1] = __builtin_amdgcn_wmma_f32_16x16x4_f32(false, a, false, b1, (short)0, acc[1], false, false);
        acc[2] = __builtin_amdgcn_wmma_f32_16x16x4_f32(false, a, false, b2, (short)0, acc[2], false, false);
        acc[3] = __builtin_amdgcn_wmma_f32_16x16x4_f32(false, a, false, b3, (short)0, acc[3], false, false);
    }

#pragma unroll
    for (int j = 0; j < 4; ++j) {
#pragma unroll
        for (int r = 0; r < 8; ++r) {
            int row = m0 + r + hh * 8;
            int col = n0 + j * 16 + (lane & 15);
            C[(size_t)row * N + col] = acc[j][r];
        }
    }
}

// ---------------------------------------------------------------------------
// scores += pos_bias; diagonal = -1e9
// ---------------------------------------------------------------------------
__global__ void scores_post_k(float* __restrict__ scores, const float* __restrict__ pos_bias)
{
    size_t idx = (size_t)blockIdx.x * blockDim.x + threadIdx.x;
    size_t total = (size_t)Bn * Pn * Pn;
    if (idx >= total) return;
    int q = (int)(idx % Pn);
    int p = (int)((idx / Pn) % Pn);
    float v = scores[idx] + pos_bias[(size_t)p * Pn + q];
    if (p == q) v = -1.0e9f;
    scores[idx] = v;
}

// ---------------------------------------------------------------------------
// Top-K (K=32) per row of P=1024 + routed softmax bias. One wave per row.
// ---------------------------------------------------------------------------
__global__ __launch_bounds__(32) void topk_route_k(
    const float* __restrict__ scores, int* __restrict__ routes, float* __restrict__ rbias)
{
    int row = blockIdx.x;              // b*P + p
    int lane = threadIdx.x;
    const float* s = scores + (size_t)row * Pn;
    float v[32];
#pragma unroll
    for (int i = 0; i < 32; ++i) v[i] = s[i * 32 + lane];

    float topv = 0.f; int topi = 0;
    for (int t = 0; t < Kn; ++t) {
        float bv = -3.0e38f; int bi = 0x7fffffff;
#pragma unroll
        for (int i = 0; i < 32; ++i) {
            if (v[i] > bv) { bv = v[i]; bi = i * 32 + lane; }
        }
        for (int off = 16; off > 0; off >>= 1) {
            float ov = __shfl_xor(bv, off);
            int   oi = __shfl_xor(bi, off);
            if (ov > bv || (ov == bv && oi < bi)) { bv = ov; bi = oi; }
        }
        if (lane == t) { topv = bv; topi = bi; }
        if (lane == (bi & 31)) v[bi >> 5] = -3.0e38f;
    }
    float t0 = __shfl(topv, 0);
    float lg = (topv - t0) * (1.0f / TEMPER);
    float e = expf(lg);
    float sum = e;
    for (int off = 16; off > 0; off >>= 1) sum += __shfl_xor(sum, off);
    float lw = lg - logf(sum);                 // log softmax weight
    routes[(size_t)row * Kn + lane] = topi;
    rbias [(size_t)row * Kn + lane] = fmaxf(lw, -10.0f);
}

// ---------------------------------------------------------------------------
// CLS attention: one block per (b,h); softmax over all S keys; writes row s=0
// ---------------------------------------------------------------------------
__global__ __launch_bounds__(128) void cls_attn_k(
    const float* __restrict__ qkv, float* __restrict__ out, f16* __restrict__ outh)
{
    int bh = blockIdx.x;
    int b = bh / Hn, h = bh % Hn;
    __shared__ float q[HDn];
    __shared__ float sbuf[Sn];
    __shared__ float red[128];
    int t = threadIdx.x;
    if (t < HDn) q[t] = qkv[(size_t)(b * Sn) * (3 * Dn) + h * HDn + t];
    __syncthreads();

    float lmax = -3.0e38f;
    for (int s = t; s < Sn; s += 128) {
        const float* kr = qkv + (size_t)(b * Sn + s) * (3 * Dn) + (Hn + h) * HDn;
        float d = 0.f;
        for (int i = 0; i < HDn; ++i) d += q[i] * kr[i];
        d *= SCALEQ;
        sbuf[s] = d;
        lmax = fmaxf(lmax, d);
    }
    red[t] = lmax; __syncthreads();
    for (int o = 64; o > 0; o >>= 1) { if (t < o) red[t] = fmaxf(red[t], red[t + o]); __syncthreads(); }
    float m = red[0]; __syncthreads();
    float lsum = 0.f;
    for (int s = t; s < Sn; s += 128) { float e = expf(sbuf[s] - m); sbuf[s] = e; lsum += e; }
    red[t] = lsum; __syncthreads();
    for (int o = 64; o > 0; o >>= 1) { if (t < o) red[t] += red[t + o]; __syncthreads(); }
    float inv = 1.0f / red[0];
    __syncthreads();
    if (t < HDn) {
        float acc = 0.f;
        for (int s = 0; s < Sn; ++s)
            acc += sbuf[s] * qkv[(size_t)(b * Sn + s) * (3 * Dn) + (2 * Hn + h) * HDn + t];
        float v = acc * inv;
        size_t o = (size_t)(b * Sn) * Dn + h * HDn + t;
        out[o] = v; outh[o] = (f16)v;
    }
}

// ---------------------------------------------------------------------------
// Routed attention: one wave per (b,h,p); lane = routed key; shuffle softmax
// ---------------------------------------------------------------------------
__global__ __launch_bounds__(256) void routed_attn_k(
    const float* __restrict__ qkv, const int* __restrict__ routes,
    const float* __restrict__ rbias, float* __restrict__ out, f16* __restrict__ outh)
{
    int wave = threadIdx.x >> 5;
    int lane = threadIdx.x & 31;
    int item = blockIdx.x * 8 + wave;           // b*H*P items
    if (item >= Bn * Hn * Pn) return;
    int p = item % Pn;
    int h = (item / Pn) % Hn;
    int b = item / (Pn * Hn);

    int   rt   = routes[(size_t)(b * Pn + p) * Kn + lane];
    float bias = rbias [(size_t)(b * Pn + p) * Kn + lane];

    const float* qr = qkv + (size_t)(b * Sn + 1 + p)  * (3 * Dn) + h * HDn;
    const float* kr = qkv + (size_t)(b * Sn + 1 + rt) * (3 * Dn) + (Hn + h) * HDn;
    float d = 0.f;
    for (int i = 0; i < HDn; ++i) d += qr[i] * kr[i];
    float sp = d * SCALEQ + bias;

    float m = sp;
    for (int off = 16; off > 0; off >>= 1) m = fmaxf(m, __shfl_xor(m, off));
    float e = expf(sp - m);
    float sum = e;
    for (int off = 16; off > 0; off >>= 1) sum += __shfl_xor(sum, off);
    float attn = e / sum;

    int d0 = lane * 2;
    float a0 = 0.f, a1 = 0.f;
    for (int k = 0; k < Kn; ++k) {
        float ak = __shfl(attn, k);
        int   rk = __shfl(rt, k);
        const float* vr = qkv + (size_t)(b * Sn + 1 + rk) * (3 * Dn) + (2 * Hn + h) * HDn;
        a0 += ak * vr[d0];
        a1 += ak * vr[d0 + 1];
    }
    size_t o = (size_t)(b * Sn + 1 + p) * Dn + h * HDn + d0;
    out[o] = a0; out[o + 1] = a1;
    outh[o] = (f16)a0; outh[o + 1] = (f16)a1;
}

// ---------------------------------------------------------------------------
// Host orchestration
// ---------------------------------------------------------------------------
extern "C" void kernel_launch(void* const* d_in, const int* in_sizes, int n_in,
                              void* d_out, int out_size, void* d_ws, size_t ws_size,
                              hipStream_t stream)
{
    const float* x      = (const float*)d_in[0];
    const float* n1w    = (const float*)d_in[1];
    const float* n1b    = (const float*)d_in[2];
    const float* rq_w   = (const float*)d_in[3];
    const float* rq_b   = (const float*)d_in[4];
    const float* rk_w   = (const float*)d_in[5];
    const float* rk_b   = (const float*)d_in[6];
    const float* posb   = (const float*)d_in[7];
    const float* qkv_w  = (const float*)d_in[8];
    const float* qkv_b  = (const float*)d_in[9];
    const float* proj_w = (const float*)d_in[10];
    const float* proj_b = (const float*)d_in[11];
    const float* n2w    = (const float*)d_in[12];
    const float* n2b    = (const float*)d_in[13];
    const float* fc1_w  = (const float*)d_in[14];
    const float* fc1_b  = (const float*)d_in[15];
    const float* fc2_w  = (const float*)d_in[16];
    const float* fc2_b  = (const float*)d_in[17];
    float* out = (float*)d_out;

    const size_t BS  = (size_t)Bn * Sn;          // 2050
    const size_t BSD = BS * Dn;
    const size_t BPD = (size_t)Bn * Pn * Dn;
    const size_t BPP = (size_t)Bn * Pn * Pn;

    char* wp = (char*)d_ws;
    auto alloc = [&](size_t bytes) -> void* {
        void* r = (void*)wp;
        wp += (bytes + 255) & ~(size_t)255;
        return r;
    };

    float* xnorm   = (float*)alloc(BSD * 4);
    f16*   xnorm_h = (f16*)  alloc(BSD * 2);
    float* qr      = (float*)alloc(BPD * 4);
    float* kr      = (float*)alloc(BPD * 4);
    float* scores  = (float*)alloc(BPP * 4);
    int*   routes  = (int*)  alloc((size_t)Bn * Pn * Kn * 4);
    float* rbias   = (float*)alloc((size_t)Bn * Pn * Kn * 4);
    float* qkvo    = (float*)alloc(BS * 3 * Dn * 4);
    float* attno   = (float*)alloc(BSD * 4);
    f16*   attno_h = (f16*)  alloc(BSD * 2);
    float* x1      = (float*)alloc(BSD * 4);
    float* hn      = (float*)alloc(BSD * 4);
    f16*   hn_h    = (f16*)  alloc(BSD * 2);
    f16*   h1_h    = (f16*)  alloc(BS * 4 * Dn * 2);
    f16*   rq_wh   = (f16*)  alloc((size_t)Dn * Dn * 2);
    f16*   rk_wh   = (f16*)  alloc((size_t)Dn * Dn * 2);
    f16*   qkv_wh  = (f16*)  alloc((size_t)3 * Dn * Dn * 2);
    f16*   proj_wh = (f16*)  alloc((size_t)Dn * Dn * 2);
    f16*   fc1_wh  = (f16*)  alloc((size_t)4 * Dn * Dn * 2);
    f16*   fc2_wh  = (f16*)  alloc((size_t)Dn * 4 * Dn * 2);

    auto cvt = [&](const float* src, f16* dst, long n) {
        int grid = (int)((n + 1023) / 1024);
        if (grid > 8192) grid = 8192;
        cvt_f16_k<<<grid, 1024, 0, stream>>>(src, dst, n);
    };
    cvt(rq_w,   rq_wh,   (long)Dn * Dn);
    cvt(rk_w,   rk_wh,   (long)Dn * Dn);
    cvt(qkv_w,  qkv_wh,  (long)3 * Dn * Dn);
    cvt(proj_w, proj_wh, (long)Dn * Dn);
    cvt(fc1_w,  fc1_wh,  (long)4 * Dn * Dn);
    cvt(fc2_w,  fc2_wh,  (long)Dn * 4 * Dn);

    // LN1
    layernorm_k<<<(int)BS, 256, 0, stream>>>(x, n1w, n1b, xnorm, xnorm_h);

    // rq / rk linear (per batch, xp = xnorm rows [b*S+1, b*S+1+P))
    for (int b = 0; b < Bn; ++b) {
        const f16* Ab = xnorm_h + (size_t)(b * Sn + 1) * Dn;
        dim3 g(Pn / 128, Dn / 64);
        wmma_gemm_f16_k<<<g, 256, 0, stream>>>(Ab, rq_wh, rq_b, nullptr,
                                               qr + (size_t)b * Pn * Dn, nullptr,
                                               Pn, Dn, Dn, 0);
        wmma_gemm_f16_k<<<g, 256, 0, stream>>>(Ab, rk_wh, rk_b, nullptr,
                                               kr + (size_t)b * Pn * Dn, nullptr,
                                               Pn, Dn, Dn, 0);
    }
    l2norm_k<<<Bn * Pn, 256, 0, stream>>>(qr);
    l2norm_k<<<Bn * Pn, 256, 0, stream>>>(kr);

    // scores = qr @ kr^T per batch (full f32 WMMA to keep top-k ordering)
    for (int b = 0; b < Bn; ++b) {
        dim3 g(Pn / 16, Pn / 64);
        wmma_gemm_f32_k<<<g, 32, 0, stream>>>(qr + (size_t)b * Pn * Dn,
                                              kr + (size_t)b * Pn * Dn,
                                              scores + (size_t)b * Pn * Pn,
                                              Pn, Pn, Dn);
    }
    {
        size_t total = BPP;
        scores_post_k<<<(int)((total + 255) / 256), 256, 0, stream>>>(scores, posb);
    }
    topk_route_k<<<Bn * Pn, 32, 0, stream>>>(scores, routes, rbias);

    // qkv projection
    {
        dim3 g((unsigned)((BS + 127) / 128), (unsigned)(3 * Dn / 64));
        wmma_gemm_f16_k<<<g, 256, 0, stream>>>(xnorm_h, qkv_wh, qkv_b, nullptr,
                                               qkvo, nullptr, (int)BS, 3 * Dn, Dn, 0);
    }

    // attention
    cls_attn_k<<<Bn * Hn, 128, 0, stream>>>(qkvo, attno, attno_h);
    routed_attn_k<<<(Bn * Hn * Pn) / 8, 256, 0, stream>>>(qkvo, routes, rbias, attno, attno_h);

    // proj + residual
    {
        dim3 g((unsigned)((BS + 127) / 128), (unsigned)(Dn / 64));
        wmma_gemm_f16_k<<<g, 256, 0, stream>>>(attno_h, proj_wh, proj_b, x,
                                               x1, nullptr, (int)BS, Dn, Dn, 0);
    }

    // LN2
    layernorm_k<<<(int)BS, 256, 0, stream>>>(x1, n2w, n2b, hn, hn_h);

    // fc1 + GELU (f16 output only)
    {
        dim3 g((unsigned)((BS + 127) / 128), (unsigned)(4 * Dn / 64));
        wmma_gemm_f16_k<<<g, 256, 0, stream>>>(hn_h, fc1_wh, fc1_b, nullptr,
                                               nullptr, h1_h, (int)BS, 4 * Dn, Dn, 1);
    }

    // fc2 + residual -> d_out
    {
        dim3 g((unsigned)((BS + 127) / 128), (unsigned)(Dn / 64));
        wmma_gemm_f16_k<<<g, 256, 0, stream>>>(h1_h, fc2_wh, fc2_b, x1,
                                               out, nullptr, (int)BS, Dn, 4 * Dn, 0);
    }

    (void)in_sizes; (void)n_in; (void)out_size; (void)ws_size;
}
